// ELRLRTModel_41695542509949
// MI455X (gfx1250) — compile-verified, matrix operand order checked by reading, not run on hardware
//
#include <hip/hip_runtime.h>
#include <hip/hip_bf16.h>

#define D_   512
#define H_   8
#define DH_  64
#define L_   6
#define DFF_ 2048
#define V_   256
#define K_   5
#define TAU2 0.01f   // TAU^2, compare against squared norm
#define EPS_ 1e-5f

typedef __attribute__((ext_vector_type(16))) __bf16 v16bf;
typedef __attribute__((ext_vector_type(8)))  __bf16 v8bf;
typedef __attribute__((ext_vector_type(4)))  __bf16 v4bf;
typedef __attribute__((ext_vector_type(8)))  float  v8f;

static __device__ __forceinline__ __bf16 tobf(float f) { return (__bf16)f; }

// Issue one 16-byte async global->LDS copy (CDNA5 GLOBAL_LOAD_ASYNC_TO_LDS_B128).
// ldsoff: LDS byte offset (generic shared ptr low 32 bits), gptr: global address.
static __device__ __forceinline__ void async_copy_b128(unsigned ldsoff, const void* gptr) {
  asm volatile("global_load_async_to_lds_b128 %0, %1, off"
               :: "v"(ldsoff), "v"(gptr) : "memory");
}
static __device__ __forceinline__ void wait_async0() {
#if __has_builtin(__builtin_amdgcn_s_wait_asynccnt)
  __builtin_amdgcn_s_wait_asynccnt(0);
#else
  asm volatile("s_wait_asynccnt 0" ::: "memory");
#endif
}

// Load a 16-bit WMMA A/B fragment from an LDS row (K-window of 32 halves at rowp).
// Per CDNA5 ISA layout: lanes 0-15 hold K 0-7 (v0-3) and K 16-23 (v4-7);
// lanes 16-31 hold K 8-15 and K 24-31.  hi = lane>>4.
static __device__ __forceinline__ v16bf load_frag(const __bf16* rowp, int hi) {
  v8bf a0 = *(const v8bf*)(rowp + hi * 8);
  v8bf a1 = *(const v8bf*)(rowp + 16 + hi * 8);
  return __builtin_shufflevector(a0, a1, 0,1,2,3,4,5,6,7,8,9,10,11,12,13,14,15);
}

// ---------------------------------------------------------------------------
// fp32 -> bf16 bulk convert (n must be a multiple of 4)
// ---------------------------------------------------------------------------
__global__ __launch_bounds__(256)
void cvt_bf16_kernel(const float* __restrict__ in, __bf16* __restrict__ out, int n)
{
  int i = (blockIdx.x * 256 + threadIdx.x) * 4;
  if (i >= n) return;
  float4 f = *(const float4*)(in + i);
  v4bf o;
  o[0] = tobf(f.x); o[1] = tobf(f.y); o[2] = tobf(f.z); o[3] = tobf(f.w);
  *(v4bf*)(out + i) = o;
}

// ---------------------------------------------------------------------------
// GEMM: C[M][N] = act( A[M][Kd] @ W[N][Kd]^T + bias[N] ),  A/W already bf16.
// Block: 256 threads (8 waves), block tile 64(M) x 128(N), K step 32.
// Tiles staged with async global->LDS B128 copies (ASYNCcnt), fragments via
// ds_load_b128, math via v_wmma_f32_16x16x32_bf16.
// ---------------------------------------------------------------------------
template<int RELU>
__global__ __launch_bounds__(256)
void gemm_bias_kernel(const __bf16* __restrict__ A, const __bf16* __restrict__ W,
                      const float* __restrict__ bias, float* __restrict__ C,
                      int M, int N, int Kd)
{
  __shared__ __attribute__((aligned(16))) __bf16 As[64][40];    // pad 32->40 halves
  __shared__ __attribute__((aligned(16))) __bf16 Ws[128][40];

  const int t    = threadIdx.x;
  const int lane = t & 31;
  const int wv   = t >> 5;
  const int hi   = lane >> 4;
  const int l16  = lane & 15;
  const int m0   = blockIdx.y * 64;
  const int n0   = blockIdx.x * 128;
  const int mt   = (wv & 3) * 16;
  const int nh   = wv >> 2;      // 0..1 -> n columns nh*64 .. nh*64+63

  // async staging map: thread t moves 16B chunk (row = t>>2, halves (t&3)*8)
  const int ar = t >> 2;          // 0..63
  const int ac = (t & 3) * 8;     // 0,8,16,24
  const unsigned ldsA  = (unsigned)(size_t)&As[ar][ac];
  const unsigned ldsW0 = (unsigned)(size_t)&Ws[ar][ac];
  const unsigned ldsW1 = (unsigned)(size_t)&Ws[ar + 64][ac];

  v8f acc[4] = {v8f{}, v8f{}, v8f{}, v8f{}};

  for (int k0 = 0; k0 < Kd; k0 += 32) {
    __syncthreads();   // previous tile fully consumed by all waves
    async_copy_b128(ldsA,  A + (size_t)(m0 + ar) * Kd + k0 + ac);
    async_copy_b128(ldsW0, W + (size_t)(n0 + ar) * Kd + k0 + ac);
    async_copy_b128(ldsW1, W + (size_t)(n0 + ar + 64) * Kd + k0 + ac);
    if (k0 + 32 < Kd) {  // pull next K slab toward L2 while we compute
      __builtin_prefetch(&A[(size_t)(m0 + ar) * Kd + k0 + 32], 0, 1);
      __builtin_prefetch(&W[(size_t)(n0 + ar) * Kd + k0 + 32], 0, 1);
    }
    wait_async0();       // this wave's async copies landed in LDS
    __syncthreads();     // all waves' copies landed

    v16bf a = load_frag(&As[mt + l16][0], hi);
    #pragma unroll
    for (int j = 0; j < 4; ++j) {
      v16bf b = load_frag(&Ws[nh * 64 + j * 16 + l16][0], hi);
      acc[j] = __builtin_amdgcn_wmma_f32_16x16x32_bf16(
          false, a, false, b, (short)0, acc[j], false, false);
    }
  }

  // Epilogue: C layout -- lane l holds column l16 of rows mt+i (+8 if hi)
  #pragma unroll
  for (int j = 0; j < 4; ++j) {
    int nc = n0 + nh * 64 + j * 16 + l16;
    float bv = bias[nc];
    #pragma unroll
    for (int i = 0; i < 8; ++i) {
      int row = m0 + mt + i + 8 * hi;
      float v = acc[j][i] + bv;
      if (RELU) v = fmaxf(v, 0.f);
      C[(size_t)row * N + nc] = v;
    }
  }
}

// ---------------------------------------------------------------------------
// Flash attention: per (64-query block, head). 256 threads / 8 waves.
// S = (Q*0.125) K^T and O += P V both via bf16 WMMA; online softmax in fp32.
// Wave w: m-tile (w&3)*16, two n-tiles (w>>2)*2 + {0,1} (for both S and O).
// ---------------------------------------------------------------------------
__global__ __launch_bounds__(256)
void flash_attn_kernel(const float* __restrict__ Q, int ldq,
                       const float* __restrict__ K, int ldk,
                       const float* __restrict__ V, int ldv,
                       float* __restrict__ O, int ldo,
                       int Pq, int Pkv)
{
  __shared__ __attribute__((aligned(16))) __bf16 Qs[64][72];
  __shared__ __attribute__((aligned(16))) __bf16 Ks[64][72];   // [key][dh]
  __shared__ __attribute__((aligned(16))) __bf16 Vt[64][72];   // [dh][key] (transposed)
  __shared__ __attribute__((aligned(16))) __bf16 Ps[64][72];   // softmax probs [q][key]
  __shared__ float Ss[64][68];                                  // raw scores
  __shared__ float rowm[64], rowl[64], rowc[64];

  const int t    = threadIdx.x;
  const int lane = t & 31;
  const int wv   = t >> 5;
  const int hi   = lane >> 4;
  const int l16  = lane & 15;
  const int q0   = blockIdx.x * 64;
  const int h    = blockIdx.y;
  const float* Qh = Q + h * DH_;
  const float* Kh = K + h * DH_;
  const float* Vh = V + h * DH_;
  const int mt = (wv & 3) * 16;
  const int nh = wv >> 2;
  const int nt0 = (nh * 2 + 0) * 16;
  const int nt1 = (nh * 2 + 1) * 16;

  const int ldr = t >> 6;   // 0..3
  const int ldc = t & 63;   // 0..63

  // stage Q once (fold 1/sqrt(DH) = 0.125)
  #pragma unroll
  for (int e = 0; e < 16; ++e) {
    int r = ldr + 4 * e;
    Qs[r][ldc] = tobf(Qh[(size_t)(q0 + r) * ldq + ldc] * 0.125f);
  }
  if (t < 64) { rowm[t] = -1e30f; rowl[t] = 0.f; }

  v8f o0 = {}, o1 = {};

  for (int k0 = 0; k0 < Pkv; k0 += 64) {
    __syncthreads();
    #pragma unroll
    for (int e = 0; e < 16; ++e) {
      int r = ldr + 4 * e;
      Ks[r][ldc] = tobf(Kh[(size_t)(k0 + r) * ldk + ldc]);
      Vt[ldc][r] = tobf(Vh[(size_t)(k0 + r) * ldv + ldc]);   // transpose on store
    }
    __syncthreads();

    // S tiles: M=64 queries x N=64 keys, K = DH = 64 (2 steps of 32)
    v8f s0 = {}, s1 = {};
    #pragma unroll
    for (int ks = 0; ks < 2; ++ks) {
      int ko = ks * 32;
      v16bf a  = load_frag(&Qs[mt + l16][ko], hi);
      v16bf b0 = load_frag(&Ks[nt0 + l16][ko], hi);
      v16bf b1 = load_frag(&Ks[nt1 + l16][ko], hi);
      s0 = __builtin_amdgcn_wmma_f32_16x16x32_bf16(false, a, false, b0, (short)0, s0, false, false);
      s1 = __builtin_amdgcn_wmma_f32_16x16x32_bf16(false, a, false, b1, (short)0, s1, false, false);
    }
    #pragma unroll
    for (int i = 0; i < 8; ++i) {
      int row = mt + i + 8 * hi;
      Ss[row][nt0 + l16] = s0[i];
      Ss[row][nt1 + l16] = s1[i];
    }
    __syncthreads();

    // online softmax (one thread per query row)
    if (t < 64) {
      float mo = rowm[t], mx = mo;
      #pragma unroll 8
      for (int n = 0; n < 64; ++n) mx = fmaxf(mx, Ss[t][n]);
      float corr = __expf(mo - mx);
      float sum = 0.f;
      #pragma unroll 8
      for (int n = 0; n < 64; ++n) {
        float p = __expf(Ss[t][n] - mx);
        Ps[t][n] = tobf(p);
        sum += p;
      }
      rowl[t] = rowl[t] * corr + sum;
      rowm[t] = mx;
      rowc[t] = corr;
    }
    __syncthreads();

    // rescale O accumulators by per-row correction
    #pragma unroll
    for (int i = 0; i < 8; ++i) {
      float c = rowc[mt + i + 8 * hi];
      o0[i] *= c;
      o1[i] *= c;
    }
    // O += P @ V  (K dim = 64 keys, 2 steps of 32)
    #pragma unroll
    for (int ks = 0; ks < 2; ++ks) {
      int ko = ks * 32;
      v16bf a  = load_frag(&Ps[mt + l16][ko], hi);
      v16bf b0 = load_frag(&Vt[nt0 + l16][ko], hi);
      v16bf b1 = load_frag(&Vt[nt1 + l16][ko], hi);
      o0 = __builtin_amdgcn_wmma_f32_16x16x32_bf16(false, a, false, b0, (short)0, o0, false, false);
      o1 = __builtin_amdgcn_wmma_f32_16x16x32_bf16(false, a, false, b1, (short)0, o1, false, false);
    }
  }

  // write O / l   (O[q][h*DH + dh])
  #pragma unroll
  for (int i = 0; i < 8; ++i) {
    int row = mt + i + 8 * hi;
    float inv = 1.f / rowl[row];
    size_t qr = (size_t)(q0 + row) * ldo + h * DH_;
    O[qr + nt0 + l16] = o0[i] * inv;
    O[qr + nt1 + l16] = o1[i] * inv;
  }
}

// ---------------------------------------------------------------------------
// y = LayerNorm(x + r) * w + b     (one wave per row, wave32 shfl reduction)
// ---------------------------------------------------------------------------
__global__ __launch_bounds__(256)
void add_ln_kernel(const float* __restrict__ x, const float* __restrict__ r,
                   const float* __restrict__ w, const float* __restrict__ b,
                   float* __restrict__ y, int rows)
{
  int wv = threadIdx.x >> 5, lane = threadIdx.x & 31;
  int row = blockIdx.x * 8 + wv;
  if (row >= rows) return;
  float v[16];
  float s = 0.f;
  #pragma unroll
  for (int j = 0; j < 16; ++j) {
    size_t idx = (size_t)row * D_ + j * 32 + lane;
    v[j] = x[idx] + r[idx];
    s += v[j];
  }
  #pragma unroll
  for (int m = 16; m >= 1; m >>= 1) s += __shfl_xor(s, m, 32);
  float mean = s * (1.f / D_);
  float vs = 0.f;
  #pragma unroll
  for (int j = 0; j < 16; ++j) { float d = v[j] - mean; vs += d * d; }
  #pragma unroll
  for (int m = 16; m >= 1; m >>= 1) vs += __shfl_xor(vs, m, 32);
  float inv = rsqrtf(vs * (1.f / D_) + EPS_);
  #pragma unroll
  for (int j = 0; j < 16; ++j) {
    int c = j * 32 + lane;
    y[(size_t)row * D_ + c] = (v[j] - mean) * inv * w[c] + b[c];
  }
}

// ---------------------------------------------------------------------------
// x = (||enc_row|| > TAU) ? alpha*enc + (1-alpha)*tok : tok
// ---------------------------------------------------------------------------
__global__ __launch_bounds__(256)
void gate_mix_kernel(const float* __restrict__ enc, const float* __restrict__ tok,
                     const float* __restrict__ alpha, float* __restrict__ x, int rows)
{
  int wv = threadIdx.x >> 5, lane = threadIdx.x & 31;
  int row = blockIdx.x * 8 + wv;
  if (row >= rows) return;
  float e[16];
  float s = 0.f;
  #pragma unroll
  for (int j = 0; j < 16; ++j) {
    e[j] = enc[(size_t)row * D_ + j * 32 + lane];
    s += e[j] * e[j];
  }
  #pragma unroll
  for (int m = 16; m >= 1; m >>= 1) s += __shfl_xor(s, m, 32);
  bool g = s > TAU2;
  float a = alpha[0];
  #pragma unroll
  for (int j = 0; j < 16; ++j) {
    size_t idx = (size_t)row * D_ + j * 32 + lane;
    float tv = tok[idx];
    x[idx] = g ? a * e[j] + (1.f - a) * tv : tv;
  }
}

// ---------------------------------------------------------------------------
// x[p][d] = mean_k emb[bytes[p*K+k]][d] ;  tok[p][d] = emb[bytes[p*K+K-1]][d]
// ---------------------------------------------------------------------------
__global__ __launch_bounds__(256)
void embed_kernel(const int* __restrict__ bytes, const float* __restrict__ emb,
                  float* __restrict__ x, float* __restrict__ tok, int P)
{
  int gid = blockIdx.x * 256 + threadIdx.x;
  if (gid >= P * D_) return;
  int p = gid / D_, d = gid - p * D_;
  float s = 0.f;
  #pragma unroll
  for (int k = 0; k < K_; ++k)
    s += emb[(size_t)bytes[p * K_ + k] * D_ + d];
  x[gid] = s * (1.f / K_);
  tok[gid] = emb[(size_t)bytes[p * K_ + K_ - 1] * D_ + d];
}

// ---------------------------------------------------------------------------
extern "C" void kernel_launch(void* const* d_in, const int* in_sizes, int n_in,
                              void* d_out, int out_size, void* d_ws, size_t ws_size,
                              hipStream_t stream) {
  (void)n_in; (void)out_size; (void)ws_size;
  const int*   bytes    = (const int*)  d_in[0];
  const float* emb      = (const float*)d_in[1];
  const float* alpha    = (const float*)d_in[2];
  const float* enc_inw  = (const float*)d_in[3];
  const float* enc_inb  = (const float*)d_in[4];
  const float* enc_ow   = (const float*)d_in[5];
  const float* enc_ob   = (const float*)d_in[6];
  const float* enc_w1   = (const float*)d_in[7];
  const float* enc_b1   = (const float*)d_in[8];
  const float* enc_w2   = (const float*)d_in[9];
  const float* enc_b2   = (const float*)d_in[10];
  const float* enc_ln1w = (const float*)d_in[11];
  const float* enc_ln1b = (const float*)d_in[12];
  const float* enc_ln2w = (const float*)d_in[13];
  const float* enc_ln2b = (const float*)d_in[14];
  const float* dsa_inw  = (const float*)d_in[15];
  const float* dsa_inb  = (const float*)d_in[16];
  const float* dsa_ow   = (const float*)d_in[17];
  const float* dsa_ob   = (const float*)d_in[18];
  const float* dca_inw  = (const float*)d_in[19];
  const float* dca_inb  = (const float*)d_in[20];
  const float* dca_ow   = (const float*)d_in[21];
  const float* dca_ob   = (const float*)d_in[22];
  const float* dec_w1   = (const float*)d_in[23];
  const float* dec_b1   = (const float*)d_in[24];
  const float* dec_w2   = (const float*)d_in[25];
  const float* dec_b2   = (const float*)d_in[26];
  const float* dln1w    = (const float*)d_in[27];
  const float* dln1b    = (const float*)d_in[28];
  const float* dln2w    = (const float*)d_in[29];
  const float* dln2b    = (const float*)d_in[30];
  const float* dln3w    = (const float*)d_in[31];
  const float* dln3b    = (const float*)d_in[32];
  const float* out_w    = (const float*)d_in[33];
  const float* out_b    = (const float*)d_in[34];

  const int P = in_sizes[0] / K_;   // 2048
  const size_t PD = (size_t)P * D_;

  float* ws  = (float*)d_ws;
  float* x   = ws;             // P*D
  float* tok = ws + PD;        // P*D
  float* enc = ws + 2 * PD;    // P*D
  float* qkv = ws + 3 * PD;    // P*3D  (also Q [P*D] + KV [P*2D] for cross-attn)
  float* tmp = ws + 6 * PD;    // P*D   (attn output / value proj)
  float* prj = ws + 7 * PD;    // P*D   (projection results / residual branch)
  float* ff  = ws + 8 * PD;    // P*DFF (= 4*PD)
  __bf16* Abf = (__bf16*)(ws + 12 * PD);  // up to P*DFF bf16 (= 2*PD floats)
  __bf16* Wbf = (__bf16*)(ws + 14 * PD);  // up to DFF*D bf16

  dim3 blk(256);

  auto GEMM = [&](const float* A, const float* W, const float* B, float* C,
                  int M, int N, int Kd, bool relu) {
    int na = M * Kd, nw = N * Kd;
    cvt_bf16_kernel<<<(na / 4 + 255) / 256, blk, 0, stream>>>(A, Abf, na);
    cvt_bf16_kernel<<<(nw / 4 + 255) / 256, blk, 0, stream>>>(W, Wbf, nw);
    dim3 g(N / 128, M / 64);
    if (relu) gemm_bias_kernel<1><<<g, blk, 0, stream>>>(Abf, Wbf, B, C, M, N, Kd);
    else      gemm_bias_kernel<0><<<g, blk, 0, stream>>>(Abf, Wbf, B, C, M, N, Kd);
  };

  embed_kernel<<<(P * D_ + 255) / 256, blk, 0, stream>>>(bytes, emb, x, tok, P);

  // -------- encoder --------
  for (int l = 0; l < L_; ++l) {
    const float* inw = enc_inw + (size_t)l * 3 * D_ * D_;
    const float* inb = enc_inb + (size_t)l * 3 * D_;
    GEMM(x, inw, inb, qkv, P, 3 * D_, D_, false);
    flash_attn_kernel<<<dim3(P / 64, H_), blk, 0, stream>>>(
        qkv, 3 * D_, qkv + D_, 3 * D_, qkv + 2 * D_, 3 * D_, tmp, D_, P, P);
    GEMM(tmp, enc_ow + (size_t)l * D_ * D_, enc_ob + l * D_, prj, P, D_, D_, false);
    add_ln_kernel<<<P / 8, blk, 0, stream>>>(x, prj, enc_ln1w + l * D_, enc_ln1b + l * D_, x, P);
    GEMM(x, enc_w1 + (size_t)l * DFF_ * D_, enc_b1 + l * DFF_, ff, P, DFF_, D_, true);
    GEMM(ff, enc_w2 + (size_t)l * D_ * DFF_, enc_b2 + l * D_, prj, P, D_, DFF_, false);
    add_ln_kernel<<<P / 8, blk, 0, stream>>>(x, prj, enc_ln2w + l * D_, enc_ln2b + l * D_, x, P);
  }
  hipMemcpyAsync(enc, x, PD * sizeof(float), hipMemcpyDeviceToDevice, stream);
  gate_mix_kernel<<<P / 8, blk, 0, stream>>>(enc, tok, alpha, x, P);

  // -------- decoder --------
  for (int l = 0; l < L_; ++l) {
    // value-only "self attention": (x @ wv^T + bv) @ ow^T + ob
    const float* sw = dsa_inw + (size_t)l * 3 * D_ * D_;
    GEMM(x, sw + 2 * (size_t)D_ * D_, dsa_inb + (size_t)l * 3 * D_ + 2 * D_, tmp, P, D_, D_, false);
    GEMM(tmp, dsa_ow + (size_t)l * D_ * D_, dsa_ob + l * D_, prj, P, D_, D_, false);
    add_ln_kernel<<<P / 8, blk, 0, stream>>>(x, prj, dln1w + l * D_, dln1b + l * D_, x, P);

    // cross attention against encoded
    const float* cw = dca_inw + (size_t)l * 3 * D_ * D_;
    const float* cb = dca_inb + (size_t)l * 3 * D_;
    GEMM(x,   cw,                      cb,       qkv,      P, D_,     D_, false);  // Q
    GEMM(enc, cw + (size_t)D_ * D_,    cb + D_,  qkv + PD, P, 2 * D_, D_, false);  // K,V
    flash_attn_kernel<<<dim3(P / 64, H_), blk, 0, stream>>>(
        qkv, D_, qkv + PD, 2 * D_, qkv + PD + D_, 2 * D_, tmp, D_, P, P);
    GEMM(tmp, dca_ow + (size_t)l * D_ * D_, dca_ob + l * D_, prj, P, D_, D_, false);
    add_ln_kernel<<<P / 8, blk, 0, stream>>>(x, prj, dln2w + l * D_, dln2b + l * D_, x, P);

    // FFN
    GEMM(x, dec_w1 + (size_t)l * DFF_ * D_, dec_b1 + l * DFF_, ff, P, DFF_, D_, true);
    GEMM(ff, dec_w2 + (size_t)l * D_ * DFF_, dec_b2 + l * D_, prj, P, D_, DFF_, false);
    add_ln_kernel<<<P / 8, blk, 0, stream>>>(x, prj, dln3w + l * D_, dln3b + l * D_, x, P);
  }

  // logits: [P, V] = x @ out_w^T + out_b
  GEMM(x, out_w, out_b, (float*)d_out, P, V_, D_, false);
}
